// DCTEnhancement_20675972563729
// MI455X (gfx1250) — matrix operand order repaired and checked by reading, not compiled, not run on hardware
//
#include <hip/hip_runtime.h>
#include <cstdint>

// ---------------------------------------------------------------------------
// DCT-II along D=256 as a WMMA GEMM: out = X(131072x256) * C^T(256x256), f32.
// bf16 split-precision (hi/lo, 3 WMMA products) on v_wmma_f32_16x16x32_bf16.
// Column-split workgroups: each WG owns 128 output columns -> 135KB LDS for
// its half of C, 64 resident accumulator VGPRs per wave, no spills.
// ---------------------------------------------------------------------------

typedef __attribute__((ext_vector_type(16))) __bf16          v16bf;
typedef __attribute__((ext_vector_type(16))) unsigned short  v16us;
typedef __attribute__((ext_vector_type(8)))  float           v8f;
typedef __attribute__((ext_vector_type(4)))  float           v4f;

#define DDIM            256
#define M_ROWS          (32 * 4096)                 // 131072 rows
#define WAVES_PER_WG    4
#define ROWS_PER_BLOCK  (WAVES_PER_WG * 16)         // 64
#define NUM_ROW_BLOCKS  (M_ROWS / ROWS_PER_BLOCK)   // 2048
#define KH              128                         // output columns per WG
#define NTILES          (KH / 16)                   // 8 col tiles per wave
#define LDS_PITCH       264                         // 528B row stride => bank skew 4
#define GRID_WGS        512                         // even: blockIdx&1 = column half

__device__ __forceinline__ unsigned short bf16_rne(float f) {
    uint32_t u = __builtin_bit_cast(uint32_t, f);
    uint32_t r = (u + 0x7FFFu + ((u >> 16) & 1u)) >> 16;
    return (unsigned short)r;
}
__device__ __forceinline__ float bf16_up(unsigned short h) {
    return __builtin_bit_cast(float, (uint32_t)h << 16);
}

__global__ void __launch_bounds__(128, 2)
dct256_wmma_bf16x3_kernel(const float* __restrict__ x, float* __restrict__ out) {
    extern __shared__ unsigned short smem[];
    unsigned short* Chi = smem;                     // [KH][LDS_PITCH] bf16 bits
    unsigned short* Clo = smem + KH * LDS_PITCH;    // [KH][LDS_PITCH] bf16 bits

    const int tid = threadIdx.x;                    // 0..127
    const int h   = blockIdx.x & 1;                 // which 128-column half

    // ---- Build this half's DCT rows in LDS, split into bf16 hi/lo. ----
    // C[k][d] = s_k * cos(pi*(2d+1)*k / 512). (2d+1)*k mod 1024 is an exact
    // integer range reduction (period 2*pi), so cosf sees a small argument.
    {
        const float pi_over_512 = 3.14159265358979323846f / 512.0f;
        const float s0 = 0.0625f;                   // sqrt(1/256)
        const float sk = 0.08838834764831845f;      // sqrt(2/256)
        for (int k = 0; k < KH; ++k) {
            const int   kg = KH * h + k;            // global DCT row
            const float sc = (kg == 0) ? s0 : sk;
#pragma unroll
            for (int dd = 0; dd < 2; ++dd) {
                const int d = tid + dd * 128;       // column 0..255
                const int t = ((2 * d + 1) * kg) & 1023;
                float val = sc * cosf((float)t * pi_over_512);
                unsigned short hb = bf16_rne(val);
                unsigned short lb = bf16_rne(val - bf16_up(hb));
                Chi[k * LDS_PITCH + d] = hb;
                Clo[k * LDS_PITCH + d] = lb;
            }
        }
    }
    __syncthreads();

    const int lane    = tid & 31;
    const int wave    = tid >> 5;                   // 0..3
    const int halfsel = lane >> 4;                  // 0: lanes 0-15, 1: 16-31
    const int l15     = lane & 15;

    for (int rb = (blockIdx.x >> 1); rb < NUM_ROW_BLOCKS; rb += (gridDim.x >> 1)) {
        const int m0 = rb * ROWS_PER_BLOCK + wave * 16;
        const float* xrow = x + (size_t)(m0 + l15) * DDIM;   // A row = lane&15

        v8f acc[NTILES];
#pragma unroll
        for (int t = 0; t < NTILES; ++t) acc[t] = (v8f){};

        // Prefetch k-step 0 of this wave's 16x256 f32 strip.
        // A in contiguous k order (same as B): lanes 0-15 hold K d0..d0+15 with
        // d0 = 32*s; lanes 16-31 hold the +16 shifted chunk.
        int d0 = halfsel * 16;
        v4f nf0 = *reinterpret_cast<const v4f*>(xrow + d0);
        v4f nf1 = *reinterpret_cast<const v4f*>(xrow + d0 + 4);
        v4f nf2 = *reinterpret_cast<const v4f*>(xrow + d0 + 8);
        v4f nf3 = *reinterpret_cast<const v4f*>(xrow + d0 + 12);

        for (int s = 0; s < 8; ++s) {
            v4f f0 = nf0, f1 = nf1, f2 = nf2, f3 = nf3;
            if (s < 7) {                            // overlap next load w/ WMMAs
                const int dn = (s + 1) * 32 + halfsel * 16;
                nf0 = *reinterpret_cast<const v4f*>(xrow + dn);
                nf1 = *reinterpret_cast<const v4f*>(xrow + dn + 4);
                nf2 = *reinterpret_cast<const v4f*>(xrow + dn + 8);
                nf3 = *reinterpret_cast<const v4f*>(xrow + dn + 12);
            }

            float fv[16];
#pragma unroll
            for (int j = 0; j < 4; ++j) {
                fv[j]      = f0[j];
                fv[4 + j]  = f1[j];
                fv[8 + j]  = f2[j];
                fv[12 + j] = f3[j];
            }
            v16us hi, lo;
#pragma unroll
            for (int j = 0; j < 16; ++j) {
                unsigned short hb = bf16_rne(fv[j]);
                hi[j] = hb;
                lo[j] = bf16_rne(fv[j] - bf16_up(hb));
            }
            const v16bf Ahi = __builtin_bit_cast(v16bf, hi);
            const v16bf Alo = __builtin_bit_cast(v16bf, lo);

            // B fragment: lane holds column t*16+l15, 16 contiguous d values.
            const int boff = s * 32 + halfsel * 16;
#pragma unroll
            for (int t = 0; t < NTILES; ++t) {
                const int r = t * 16 + l15;         // LDS row (local DCT row)
                v16bf Bhi = __builtin_bit_cast(
                    v16bf, *reinterpret_cast<const v16us*>(Chi + r * LDS_PITCH + boff));
                v16bf Blo = __builtin_bit_cast(
                    v16bf, *reinterpret_cast<const v16us*>(Clo + r * LDS_PITCH + boff));
                // hi*hi + lo*hi + hi*lo (lo*lo below f32 ulp)
                acc[t] = __builtin_amdgcn_wmma_f32_16x16x32_bf16(
                    false, Ahi, false, Bhi, (short)0, acc[t], false, false);
                acc[t] = __builtin_amdgcn_wmma_f32_16x16x32_bf16(
                    false, Alo, false, Bhi, (short)0, acc[t], false, false);
                acc[t] = __builtin_amdgcn_wmma_f32_16x16x32_bf16(
                    false, Ahi, false, Blo, (short)0, acc[t], false, false);
            }
        }

        // D layout: VGPR r -> row m0 + 8*halfsel + r, col = h*128 + t*16 + l15.
#pragma unroll
        for (int t = 0; t < NTILES; ++t) {
            float* orow = out + (size_t)(m0 + halfsel * 8) * DDIM
                              + h * KH + t * 16 + l15;
#pragma unroll
            for (int r = 0; r < 8; ++r) {
                orow[(size_t)r * DDIM] = acc[t][r];
            }
        }
    }
}

extern "C" void kernel_launch(void* const* d_in, const int* in_sizes, int n_in,
                              void* d_out, int out_size, void* d_ws, size_t ws_size,
                              hipStream_t stream) {
    (void)in_sizes; (void)n_in; (void)d_ws; (void)ws_size; (void)out_size;
    const float* x = (const float*)d_in[0];
    float* out = (float*)d_out;

    const size_t shmem = (size_t)2 * KH * LDS_PITCH * sizeof(unsigned short); // 135168 B
    hipFuncSetAttribute((const void*)dct256_wmma_bf16x3_kernel,
                        hipFuncAttributeMaxDynamicSharedMemorySize, (int)shmem);

    // 512 workgroups: blockIdx&1 selects column half, rest grid-strides over
    // 2048 row blocks of 64 rows.
    dct256_wmma_bf16x3_kernel<<<GRID_WGS, 128, shmem, stream>>>(x, out);
}
